// HRRRouter_84181359001985
// MI455X (gfx1250) — matrix operand order repaired and checked by reading, not compiled, not run on hardware
//
#include <hip/hip_runtime.h>
#include <hip/hip_bf16.h>

// ---------------------------------------------------------------------------
// HRR router: x[N,H] @ W[D,H]^T  -> circulant unbind -> cosine scores -> top2.
// Normalization of xp cancels in the cosine; numerator collapses to xpu @ M
// (M = C E^T, [D,8]); only the denominator ||xpu C|| needs the second big GEMM,
// reduced on the fly to per-row sum-of-squares (S never materialized).
// ---------------------------------------------------------------------------
#define NN 16384
#define HH 2048
#define DD 2048
#define KK 8

typedef __bf16 v16bf __attribute__((ext_vector_type(16)));
typedef float  v8f   __attribute__((ext_vector_type(8)));
typedef int    v4i   __attribute__((ext_vector_type(4)));

#if __has_builtin(__builtin_amdgcn_global_load_async_to_lds_b128)
#define HAVE_ASYNC_LDS 1
#endif

typedef __attribute__((address_space(1))) v4i GV4;   // global int4
typedef __attribute__((address_space(3))) v4i LV4;   // LDS int4

// ---- helpers ---------------------------------------------------------------

// pack two f32 -> one dword of 2x bf16 via the native pack instruction
__device__ inline unsigned pk2(float a, float b) {
#if defined(__gfx1250__)
  unsigned r;
  asm("v_cvt_pk_bf16_f32 %0, %1, %2" : "=v"(r) : "v"(a), "v"(b));
  return r;
#else
  __bf16 ha = (__bf16)a, hb = (__bf16)b;
  unsigned short ua, ub;
  __builtin_memcpy(&ua, &ha, 2);
  __builtin_memcpy(&ub, &hb, 2);
  return ((unsigned)ub << 16) | (unsigned)ua;
#endif
}
__device__ inline uint4 pack8(float4 a, float4 b) {
  return make_uint4(pk2(a.x, a.y), pk2(a.z, a.w), pk2(b.x, b.y), pk2(b.z, b.w));
}

// 16B global -> LDS copy, async if the toolchain exposes the CDNA5 builtin.
__device__ inline void cp16_async(const __bf16* g, __bf16* l) {
#ifdef HAVE_ASYNC_LDS
  __builtin_amdgcn_global_load_async_to_lds_b128(
      (GV4*)(size_t)(const void*)g,
      (LV4*)(unsigned)(size_t)(void*)l, 0, 0);
#else
  *(uint4*)l = *(const uint4*)g;
#endif
}
__device__ inline void wait_async0() {
#if __has_builtin(__builtin_amdgcn_s_wait_asynccnt)
  __builtin_amdgcn_s_wait_asynccnt(0);
#else
  asm volatile("s_wait_asynccnt 0x0" ::: "memory");
#endif
}

// A fragment (16x32 bf16): lanes 0-15 row M=lane, K {0..7,16..23};
// lanes 16-31 row M=lane-16, K {8..15,24..31}. LDS row-major stride 32.
__device__ inline v16bf load_frag_a(const __bf16* sA, int lane) {
  const int row = lane & 15;
  const int k0  = (lane >> 4) << 3;
  const uint4* p = (const uint4*)sA;
  union { uint4 u[2]; v16bf v; } f;
  f.u[0] = p[(row * 32 + k0) >> 3];
  f.u[1] = p[(row * 32 + 16 + k0) >> 3];
  return f.v;
}
// B fragment (32x16 bf16), staged transposed as [n][k] stride 32.
__device__ inline v16bf load_frag_b(const __bf16* sB, int lane) {
  const int n  = lane & 15;
  const int k0 = (lane >> 4) << 3;
  const uint4* p = (const uint4*)sB;
  union { uint4 u[2]; v16bf v; } f;
  const int base = (n * 32 + k0) >> 3;
  f.u[0] = p[base];
  f.u[1] = p[base + 2];
  return f.v;
}
__device__ inline v8f wmma_bf16(v16bf a, v16bf b, v8f c) {
  return __builtin_amdgcn_wmma_f32_16x16x32_bf16(false, a, false, b,
                                                 (short)0, c, false, false);
}

// ---- prep kernels ----------------------------------------------------------

__global__ __launch_bounds__(256) void k_prep(const float* __restrict__ lab,
                                              const float* __restrict__ sig,
                                              float* __restrict__ L,
                                              float* __restrict__ E) {
  const int k = blockIdx.x;
  const int t = threadIdx.x;
  __shared__ float red[256];

  float p = 0.f;
  for (int j = t; j < DD; j += 256) { float v = lab[k * DD + j]; p += v * v; }
  red[t] = p; __syncthreads();
  for (int s = 128; s > 0; s >>= 1) { if (t < s) red[t] += red[t + s]; __syncthreads(); }
  float inv = 1.0f / fmaxf(sqrtf(red[0]), 1e-12f);
  __syncthreads();
  for (int j = t; j < DD; j += 256) L[k * DD + j] = lab[k * DD + j] * inv;

  p = 0.f;
  for (int j = t; j < DD; j += 256) { float v = sig[k * DD + j]; p += v * v; }
  red[t] = p; __syncthreads();
  for (int s = 128; s > 0; s >>= 1) { if (t < s) red[t] += red[t + s]; __syncthreads(); }
  inv = 1.0f / fmaxf(sqrtf(red[0]), 1e-12f);
  __syncthreads();
  for (int j = t; j < DD; j += 256) E[k * DD + j] = sig[k * DD + j] * inv;
}

// R[d] = sum_k sum_j E[k,j] * L[k,(d-j) mod D]
__global__ __launch_bounds__(256) void k_R(const float* __restrict__ L,
                                           const float* __restrict__ E,
                                           float* __restrict__ R) {
  const int d = blockIdx.x * 256 + threadIdx.x;
  float acc = 0.f;
  for (int k = 0; k < KK; ++k) {
    const float* e = E + k * DD;
    const float* l = L + k * DD;
    for (int j = 0; j < DD; ++j) acc += e[j] * l[(d - j) & (DD - 1)];
  }
  R[d] = acc;
}

// M[m,k] = sum_t R[(m-t) mod D] * E[k,t]
__global__ __launch_bounds__(256) void k_M(const float* __restrict__ R,
                                           const float* __restrict__ E,
                                           float* __restrict__ M) {
  const int idx = blockIdx.x * 256 + threadIdx.x;
  const int m = idx >> 3;
  const int k = idx & 7;
  const float* e = E + k * DD;
  float acc = 0.f;
  for (int t = 0; t < DD; ++t) acc += R[(m - t) & (DD - 1)] * e[t];
  M[m * KK + k] = acc;
}

// Ct[t*D + m] = C[m,t] = R[(m-t) mod D] in bf16
__global__ __launch_bounds__(256) void k_Ct(const float* __restrict__ R,
                                            __bf16* __restrict__ Ct) {
  const int idx = blockIdx.x * 256 + threadIdx.x;
  const int t = idx >> 11;
  const int m = idx & (DD - 1);
  Ct[idx] = (__bf16)R[(m - t) & (DD - 1)];
}

__global__ __launch_bounds__(256) void k_zero(float* __restrict__ p) {
  p[blockIdx.x * 256 + threadIdx.x] = 0.f;
}

// bulk f32 -> bf16 conversion: 8 elements / thread, b128 in / b128 out
__global__ __launch_bounds__(256) void k_cvt(const float* __restrict__ src,
                                             __bf16* __restrict__ dst) {
  const size_t i8 = (size_t)(blockIdx.x * 256 + threadIdx.x) * 8;
  float4 a = ((const float4*)(src + i8))[0];
  float4 b = ((const float4*)(src + i8))[1];
  *(uint4*)(dst + i8) = pack8(a, b);
}

// ---- GEMM core: 128x64 tile, 8 waves, async double-buffered bf16 staging ---
// Computes acc[j] (j = col subtile 0..3) for rows [row0 + wave*16, +16).
__device__ inline void gemm_core(const __bf16* __restrict__ gAbase,
                                 const __bf16* __restrict__ gBbase,
                                 __bf16* sA0, __bf16* sA1,
                                 __bf16* sB0, __bf16* sB1,
                                 int t, int wave, int lane, v8f acc[4]) {
  const int ar = t >> 1, ak = (t & 1) << 4;     // 32B of A per thread
  const int br = t >> 2, bk = (t & 3) << 3;     // 16B of B per thread
  const __bf16* gA = gAbase + (size_t)ar * HH + ak;
  const __bf16* gB = gBbase + (size_t)br * HH + bk;
  __bf16* sA[2] = {sA0, sA1};
  __bf16* sB[2] = {sB0, sB1};

  cp16_async(gA, &sA[0][ar * 32 + ak]);
  cp16_async(gA + 8, &sA[0][ar * 32 + ak + 8]);
  cp16_async(gB, &sB[0][br * 32 + bk]);

  for (int k0 = 0; k0 < HH; k0 += 32) {
    const int buf = (k0 >> 5) & 1;
    wait_async0();
    __syncthreads();
    if (k0 + 32 < HH) {
      const int nb = buf ^ 1;
      cp16_async(gA + k0 + 32, &sA[nb][ar * 32 + ak]);
      cp16_async(gA + k0 + 40, &sA[nb][ar * 32 + ak + 8]);
      cp16_async(gB + k0 + 32, &sB[nb][br * 32 + bk]);
    }
    v16bf fa = load_frag_a(&sA[buf][wave * 16 * 32], lane);
#pragma unroll
    for (int j = 0; j < 4; ++j) {
      v16bf fb = load_frag_b(&sB[buf][j * 16 * 32], lane);
      acc[j] = wmma_bf16(fa, fb, acc[j]);
    }
  }
}

// ---- GEMM 1 (pure bf16 path): xpu = Xb @ Wb^T -------------------------------
__global__ __launch_bounds__(256) void k_gemm1b(const __bf16* __restrict__ Xb,
                                                const __bf16* __restrict__ Wb,
                                                __bf16* __restrict__ xpu) {
  __shared__ __bf16 sA[2][128 * 32];
  __shared__ __bf16 sB[2][64 * 32];
  const int row0 = blockIdx.y * 128;
  const int col0 = blockIdx.x * 64;
  const int t = threadIdx.x, wave = t >> 5, lane = t & 31;

  v8f acc[4] = {};
  gemm_core(Xb + (size_t)row0 * HH, Wb + (size_t)col0 * HH,
            sA[0], sA[1], sB[0], sB[1], t, wave, lane, acc);

  const int rb = row0 + wave * 16 + ((lane >> 4) << 3);
  const int cc = col0 + (lane & 15);
#pragma unroll
  for (int j = 0; j < 4; ++j)
#pragma unroll
    for (int v = 0; v < 8; ++v)
      xpu[(size_t)(rb + v) * DD + cc + j * 16] = (__bf16)acc[j][v];
}

// ---- GEMM 1 (fallback, fused f32->bf16 conversion in staging) --------------
__global__ __launch_bounds__(256) void k_gemm1f(const float* __restrict__ x,
                                                const float* __restrict__ w,
                                                __bf16* __restrict__ xpu) {
  __shared__ __bf16 sA[128 * 32];
  __shared__ __bf16 sB[64 * 32];
  const int row0 = blockIdx.y * 128;
  const int col0 = blockIdx.x * 64;
  const int t = threadIdx.x, wave = t >> 5, lane = t & 31;

  const int ar = t >> 1, ak = (t & 1) << 4;
  const int br = t >> 2, bk = (t & 3) << 3;
  const float* gA = x + (size_t)(row0 + ar) * HH + ak;
  const float* gB = w + (size_t)(col0 + br) * HH + bk;

  float4 a0 = ((const float4*)gA)[0], a1 = ((const float4*)gA)[1];
  float4 a2 = ((const float4*)gA)[2], a3 = ((const float4*)gA)[3];
  float4 b0 = ((const float4*)gB)[0], b1 = ((const float4*)gB)[1];

  v8f acc[4] = {};
  for (int k0 = 0; k0 < HH; k0 += 32) {
    ((uint4*)sA)[(ar * 32 + ak) >> 3]       = pack8(a0, a1);
    ((uint4*)sA)[((ar * 32 + ak) >> 3) + 1] = pack8(a2, a3);
    ((uint4*)sB)[(br * 32 + bk) >> 3]       = pack8(b0, b1);
    __syncthreads();

    if (k0 + 32 < HH) {
      const float* pa = gA + k0 + 32;
      a0 = ((const float4*)pa)[0]; a1 = ((const float4*)pa)[1];
      a2 = ((const float4*)pa)[2]; a3 = ((const float4*)pa)[3];
      const float* pb = gB + k0 + 32;
      b0 = ((const float4*)pb)[0]; b1 = ((const float4*)pb)[1];
    }

    v16bf fa = load_frag_a(sA + wave * 16 * 32, lane);
#pragma unroll
    for (int j = 0; j < 4; ++j) {
      v16bf fb = load_frag_b(sB + j * 16 * 32, lane);
      acc[j] = wmma_bf16(fa, fb, acc[j]);
    }
    __syncthreads();
  }

  const int rb = row0 + wave * 16 + ((lane >> 4) << 3);
  const int cc = col0 + (lane & 15);
#pragma unroll
  for (int j = 0; j < 4; ++j)
#pragma unroll
    for (int v = 0; v < 8; ++v)
      xpu[(size_t)(rb + v) * DD + cc + j * 16] = (__bf16)acc[j][v];
}

// ---- GEMM 2: rowsumsq(xpu @ C) ---------------------------------------------
__global__ __launch_bounds__(256) void k_gemm2(const __bf16* __restrict__ xpu,
                                               const __bf16* __restrict__ Ct,
                                               float* __restrict__ dsq) {
  __shared__ __bf16 sA[2][128 * 32];
  __shared__ __bf16 sB[2][64 * 32];
  const int row0 = blockIdx.y * 128;
  const int col0 = blockIdx.x * 64;
  const int t = threadIdx.x, wave = t >> 5, lane = t & 31;

  v8f acc[4] = {};
  gemm_core(xpu + (size_t)row0 * DD, Ct + (size_t)col0 * DD,
            sA[0], sA[1], sB[0], sB[1], t, wave, lane, acc);

  float rs[8];
#pragma unroll
  for (int v = 0; v < 8; ++v)
    rs[v] = acc[0][v] * acc[0][v] + acc[1][v] * acc[1][v] +
            acc[2][v] * acc[2][v] + acc[3][v] * acc[3][v];
#pragma unroll
  for (int m = 1; m < 16; m <<= 1)
#pragma unroll
    for (int v = 0; v < 8; ++v) rs[v] += __shfl_xor(rs[v], m, 32);

  if ((lane & 15) == 0) {
    const int rb = row0 + wave * 16 + ((lane >> 4) << 3);
#pragma unroll
    for (int v = 0; v < 8; ++v) atomicAdd(&dsq[rb + v], rs[v]);
  }
}

// ---- scoring ---------------------------------------------------------------
__global__ __launch_bounds__(256) void k_score(const __bf16* __restrict__ xpu,
                                               const float* __restrict__ M,
                                               const float* __restrict__ dsq,
                                               float* __restrict__ w_out,
                                               float* __restrict__ i_out,
                                               float* __restrict__ s_out) {
  const int wave = threadIdx.x >> 5, lane = threadIdx.x & 31;
  const int n = blockIdx.x * 8 + wave;
  const __bf16* xr = xpu + (size_t)n * DD;

  float num[KK];
#pragma unroll
  for (int k = 0; k < KK; ++k) num[k] = 0.f;
  for (int j = lane; j < DD; j += 32) {
    const float a = (float)xr[j];
    const float* Mr = M + j * KK;
#pragma unroll
    for (int k = 0; k < KK; ++k) num[k] += a * Mr[k];
  }
#pragma unroll
  for (int m = 1; m < 32; m <<= 1)
#pragma unroll
    for (int k = 0; k < KK; ++k) num[k] += __shfl_xor(num[k], m, 32);

  const float rinv = rsqrtf(fmaxf(dsq[n], 1e-24f));
  if (lane < KK) s_out[n * KK + lane] = num[lane] * rinv;

  if (lane == 0) {
    float s[KK];
#pragma unroll
    for (int k = 0; k < KK; ++k) s[k] = num[k] * rinv;
    int i1 = 0;
#pragma unroll
    for (int k = 1; k < KK; ++k) if (s[k] > s[i1]) i1 = k;
    int i2 = (i1 == 0) ? 1 : 0;
#pragma unroll
    for (int k = 0; k < KK; ++k) if (k != i1 && s[k] > s[i2]) i2 = k;
    const float eb = __expf(s[i2] - s[i1]);
    const float z = 1.f / (1.f + eb);
    w_out[n * 2 + 0] = z;
    w_out[n * 2 + 1] = eb * z;
    i_out[n * 2 + 0] = (float)i1;
    i_out[n * 2 + 1] = (float)i2;
  }
}

// ---------------------------------------------------------------------------

extern "C" void kernel_launch(void* const* d_in, const int* in_sizes, int n_in,
                              void* d_out, int out_size, void* d_ws, size_t ws_size,
                              hipStream_t stream) {
  const float* x      = (const float*)d_in[0];
  const float* proj_w = (const float*)d_in[1];
  const float* labels = (const float*)d_in[2];
  const float* sigs   = (const float*)d_in[3];

  float* L    = (float*)d_ws;
  float* E    = L + KK * DD;
  float* R    = E + KK * DD;
  float* M    = R + DD;
  float* dsq  = M + DD * KK;
  __bf16* Ct  = (__bf16*)(dsq + NN);             //  8 MB
  __bf16* xpu = Ct + (size_t)DD * DD;            // 64 MB
  __bf16* Xb  = xpu + (size_t)NN * DD;           // 64 MB (optional)
  __bf16* Wb  = Xb + (size_t)NN * HH;            //  8 MB (optional)

  const size_t need_big =
      (size_t)(KK * DD * 2 + DD + DD * KK + NN) * 4 +
      ((size_t)DD * DD + (size_t)NN * DD + (size_t)NN * HH + (size_t)DD * HH) * 2;

  float* out   = (float*)d_out;
  float* w_out = out;
  float* i_out = out + NN * 2;
  float* s_out = out + NN * 4;

  k_prep<<<KK, 256, 0, stream>>>(labels, sigs, L, E);
  k_R   <<<DD / 256, 256, 0, stream>>>(L, E, R);
  k_M   <<<(DD * KK) / 256, 256, 0, stream>>>(R, E, M);
  k_Ct  <<<(DD * DD) / 256, 256, 0, stream>>>(R, Ct);
  k_zero<<<NN / 256, 256, 0, stream>>>(dsq);

  dim3 g(DD / 64, NN / 128);                     // (32, 128)
  if (ws_size >= need_big) {
    k_cvt<<<(NN * HH) / 2048, 256, 0, stream>>>(x, Xb);
    k_cvt<<<(DD * HH) / 2048, 256, 0, stream>>>(proj_w, Wb);
    k_gemm1b<<<g, 256, 0, stream>>>(Xb, Wb, xpu);
  } else {
    k_gemm1f<<<g, 256, 0, stream>>>(x, proj_w, xpu);
  }
  k_gemm2<<<g, 256, 0, stream>>>(xpu, Ct, dsq);

  k_score<<<NN / 8, 256, 0, stream>>>(xpu, M, dsq, w_out, i_out, s_out);
}
